// MatcherSinkhorn_73538430042937
// MI455X (gfx1250) — matrix operand order repaired and checked by reading, not compiled, not run on hardware
//
#include <hip/hip_runtime.h>
#include <hip/hip_bf16.h>
#include <math.h>

// ---------------------------------------------------------------------------
// MatcherSinkhorn for MI455X (gfx1250, wave32, WMMA)
// B=16, Q=1024, T=1024, D=256, eps=0.1, 5 Sinkhorn iterations.
// ---------------------------------------------------------------------------

typedef __attribute__((ext_vector_type(16))) __bf16 v16bf;
typedef __attribute__((ext_vector_type(8)))  float  v8f;

union Frag32 {
    uint4 u[2];
    v16bf v;
};

__device__ __forceinline__ v16bf ld_frag2(const __bf16* p0, const __bf16* p1) {
    Frag32 f;
    f.u[0] = *(const uint4*)p0;
    f.u[1] = *(const uint4*)p1;
    return f.v;
}

// -------------------------- row sum-of-squares ------------------------------
__global__ __launch_bounds__(256) void norms_k(const float* __restrict__ x,
                                               float* __restrict__ out) {
    __shared__ float red[256];
    const int row = blockIdx.x;
    const float v = x[(size_t)row * 256 + threadIdx.x];
    red[threadIdx.x] = v * v;
    __syncthreads();
    for (int off = 128; off > 0; off >>= 1) {
        if (threadIdx.x < off) red[threadIdx.x] += red[threadIdx.x + off];
        __syncthreads();
    }
    if (threadIdx.x == 0) out[row] = red[0];
}

// -------------------------- workspace init ----------------------------------
__global__ void init_k(float* __restrict__ v, unsigned* __restrict__ gmaxBits, int n) {
    const int i = blockIdx.x * 256 + threadIdx.x;
    if (i < n) v[i] = 0.0f;
    if (i == 0) *gmaxBits = 0u;
}

// ------------- cost = sqrt(max(x2 + y2 - 2*x.y, 0)) via 3x-bf16 WMMA --------
__global__ __launch_bounds__(256) void cost_wmma_k(
    const float* __restrict__ preds, const float* __restrict__ targets,
    const float* __restrict__ x2, const float* __restrict__ y2,
    float* __restrict__ cost, unsigned* __restrict__ gmaxBits)
{
    constexpr int D = 256, Qn = 1024, Tn = 1024;
    constexpr int LDK = 40;  // padded row stride (80B, multiple of 16B)
    __shared__ __align__(16) __bf16 sAhi[128][LDK];
    __shared__ __align__(16) __bf16 sAlo[128][LDK];
    __shared__ __align__(16) __bf16 sBhi[128][LDK];
    __shared__ __align__(16) __bf16 sBlo[128][LDK];
    __shared__ float sMax[256];

    const int tid  = threadIdx.x;
    const int lane = tid & 31;
    const int wave = tid >> 5;
    const int wq   = wave & 3;   // 4 waves cover 128 q-rows (32 each)
    const int wt   = wave >> 2;  // 2 waves cover 128 t-cols (64 each)
    const int b    = blockIdx.z;
    const int q0   = blockIdx.y * 128;
    const int t0   = blockIdx.x * 128;

    const float* pA = preds   + ((size_t)b * Qn + q0) * D;
    const float* pB = targets + ((size_t)b * Tn + t0) * D;

    v8f acc[2][4];
    #pragma unroll
    for (int i = 0; i < 2; i++)
        #pragma unroll
        for (int j = 0; j < 4; j++)
            acc[i][j] = (v8f){0.f, 0.f, 0.f, 0.f, 0.f, 0.f, 0.f, 0.f};

    for (int kk = 0; kk < D; kk += 32) {
        // cooperative load + hi/lo bf16 split into LDS
        #pragma unroll
        for (int e = tid; e < 128 * 32; e += 256) {
            const int r = e >> 5, c = e & 31;
            float xa = pA[(size_t)r * D + kk + c];
            __bf16 ha = (__bf16)xa;
            sAhi[r][c] = ha;
            sAlo[r][c] = (__bf16)(xa - (float)ha);
            float xb = pB[(size_t)r * D + kk + c];
            __bf16 hb = (__bf16)xb;
            sBhi[r][c] = hb;
            sBlo[r][c] = (__bf16)(xb - (float)hb);
        }
        __syncthreads();

        // Fragment assembly per ISA 7.12.2 (wave32, 16-bit A 16x32, B 32x16)
        const int m  = lane & 15;
        const int ks = lane >> 4;
        v16bf ah[2], al[2], bh[4], bl[4];
        #pragma unroll
        for (int i = 0; i < 2; i++) {
            const __bf16* rh = &sAhi[wq * 32 + i * 16 + m][0];
            const __bf16* rl = &sAlo[wq * 32 + i * 16 + m][0];
            ah[i] = ld_frag2(rh + ks * 8, rh + ks * 8 + 16);
            al[i] = ld_frag2(rl + ks * 8, rl + ks * 8 + 16);
        }
        #pragma unroll
        for (int j = 0; j < 4; j++) {
            const __bf16* rh = &sBhi[wt * 64 + j * 16 + m][0];
            const __bf16* rl = &sBlo[wt * 64 + j * 16 + m][0];
            bh[j] = ld_frag2(rh + ks * 8, rh + ks * 8 + 16);
            bl[j] = ld_frag2(rl + ks * 8, rl + ks * 8 + 16);
        }

        // compensated product: x.y ~= lo*hi + hi*lo + hi*hi (small terms first)
        #pragma unroll
        for (int i = 0; i < 2; i++)
            #pragma unroll
            for (int j = 0; j < 4; j++) {
                v8f c0 = acc[i][j];
                c0 = __builtin_amdgcn_wmma_f32_16x16x32_bf16(false, al[i], false, bh[j], (short)0, c0, false, false);
                c0 = __builtin_amdgcn_wmma_f32_16x16x32_bf16(false, ah[i], false, bl[j], (short)0, c0, false, false);
                c0 = __builtin_amdgcn_wmma_f32_16x16x32_bf16(false, ah[i], false, bh[j], (short)0, c0, false, false);
                acc[i][j] = c0;
            }
        __syncthreads();
    }

    // epilogue: distances + tile max
    float lmax = 0.0f;
    const int half = lane >> 4;
    const int ncol = lane & 15;
    #pragma unroll
    for (int i = 0; i < 2; i++)
        #pragma unroll
        for (int j = 0; j < 4; j++)
            #pragma unroll
            for (int r = 0; r < 8; r++) {
                const int qg = q0 + wq * 32 + i * 16 + r + 8 * half;
                const int tg = t0 + wt * 64 + j * 16 + ncol;
                const float xy = acc[i][j][r];
                float cc = x2[b * Qn + qg] + y2[b * Tn + tg] - 2.0f * xy;
                cc = sqrtf(fmaxf(cc, 0.0f));
                cost[((size_t)b * Qn + qg) * Tn + tg] = cc;
                lmax = fmaxf(lmax, cc);
            }
    sMax[tid] = lmax;
    __syncthreads();
    for (int off = 128; off > 0; off >>= 1) {
        if (tid < off) sMax[tid] = fmaxf(sMax[tid], sMax[tid + off]);
        __syncthreads();
    }
    // cost >= 0 -> float bits order-preserving under unsigned max
    if (tid == 0) atomicMax(gmaxBits, __float_as_uint(sMax[0]));
}

// ------------------ u[b,q] = -eps*LSE_t((v - cost')/eps) --------------------
__global__ __launch_bounds__(256) void u_update_k(
    const float* __restrict__ cost, const float* __restrict__ v,
    const float* __restrict__ filt, const unsigned* __restrict__ gmaxBits,
    float* __restrict__ u)
{
    constexpr int Qn = 1024, Tn = 1024;
    __shared__ float sm[256], ss[256];
    const int q = blockIdx.x, b = blockIdx.y, tid = threadIdx.x;
    const float fc = 2.0f * __uint_as_float(*gmaxBits);
    const float* row = cost + ((size_t)b * Qn + q) * Tn;
    const float* fr  = filt + (size_t)b * Tn;
    const float* vr  = v + (size_t)b * Tn;
    float m = -INFINITY, s = 0.0f;
    for (int t = tid; t < Tn; t += 256) {
        float c = row[t];
        if (fr[t] <= 0.0f) c = fc;
        const float x = (vr[t] - c) * 10.0f;  // 1/eps = 10
        if (x > m) { s = s * expf(m - x) + 1.0f; m = x; }
        else       { s += expf(x - m); }
    }
    sm[tid] = m; ss[tid] = s;
    __syncthreads();
    for (int off = 128; off > 0; off >>= 1) {
        if (tid < off) {
            const float m1 = sm[tid], s1 = ss[tid];
            const float m2 = sm[tid + off], s2 = ss[tid + off];
            const float M = fmaxf(m1, m2);
            ss[tid] = s1 * expf(m1 - M) + s2 * expf(m2 - M);
            sm[tid] = M;
        }
        __syncthreads();
    }
    if (tid == 0) u[(size_t)b * Qn + q] = -0.1f * (sm[0] + logf(ss[0]));
}

// ------------------ v[b,t] = -eps*LSE_q((u - cost')/eps) --------------------
// 64 columns x 4 q-slices per block: column reads stay coalesced.
__global__ __launch_bounds__(256) void v_update_k(
    const float* __restrict__ cost, const float* __restrict__ u,
    const float* __restrict__ filt, const unsigned* __restrict__ gmaxBits,
    float* __restrict__ v)
{
    constexpr int Qn = 1024, Tn = 1024;
    __shared__ float sm[256], ss[256];
    const int tid = threadIdx.x;
    const int tx = tid & 63, ty = tid >> 6;
    const int b = blockIdx.y;
    const int t = blockIdx.x * 64 + tx;
    const float fc = 2.0f * __uint_as_float(*gmaxBits);
    const bool filtered = (filt[(size_t)b * Tn + t] <= 0.0f);
    const float* ur = u + (size_t)b * Qn;
    float m = -INFINITY, s = 0.0f;
    for (int q = ty; q < Qn; q += 4) {
        float c = cost[((size_t)b * Qn + q) * Tn + t];
        if (filtered) c = fc;
        const float x = (ur[q] - c) * 10.0f;
        if (x > m) { s = s * expf(m - x) + 1.0f; m = x; }
        else       { s += expf(x - m); }
    }
    sm[tid] = m; ss[tid] = s;
    __syncthreads();
    if (ty == 0) {
        float M = m, S = s;
        #pragma unroll
        for (int k = 1; k < 4; k++) {
            const float m2 = sm[tid + k * 64], s2 = ss[tid + k * 64];
            const float Mn = fmaxf(M, m2);
            S = S * expf(M - Mn) + s2 * expf(m2 - Mn);
            M = Mn;
        }
        v[(size_t)b * Tn + t] = -0.1f * (M + logf(S));
    }
}

// --------------- P = exp((u+v-cost')/eps), idx = argmax_t P -----------------
__global__ __launch_bounds__(256) void p_out_k(
    const float* __restrict__ cost, const float* __restrict__ u,
    const float* __restrict__ v, const float* __restrict__ filt,
    const unsigned* __restrict__ gmaxBits,
    float* __restrict__ outIdx, float* __restrict__ outP)
{
    constexpr int Qn = 1024, Tn = 1024;
    __shared__ float pv[256];
    __shared__ int   pi[256];
    const int q = blockIdx.x, b = blockIdx.y, tid = threadIdx.x;
    const float fc = 2.0f * __uint_as_float(*gmaxBits);
    const size_t rowOff = ((size_t)b * Qn + q) * Tn;
    const float uu = u[(size_t)b * Qn + q];
    const float* fr = filt + (size_t)b * Tn;
    const float* vr = v + (size_t)b * Tn;
    float best = -INFINITY;
    int bidx = 0;
    for (int t = tid; t < Tn; t += 256) {
        float c = cost[rowOff + t];
        if (fr[t] <= 0.0f) c = fc;
        const float pre = (uu + vr[t] - c) * 10.0f;  // monotone in P
        outP[rowOff + t] = expf(pre);
        if (pre > best) { best = pre; bidx = t; }
    }
    pv[tid] = best; pi[tid] = bidx;
    __syncthreads();
    for (int off = 128; off > 0; off >>= 1) {
        if (tid < off) {
            const float v2 = pv[tid + off];
            const int   i2 = pi[tid + off];
            if (v2 > pv[tid] || (v2 == pv[tid] && i2 < pi[tid])) {
                pv[tid] = v2; pi[tid] = i2;
            }
        }
        __syncthreads();
    }
    if (tid == 0) outIdx[(size_t)b * Qn + q] = (float)pi[0];
}

// ---------------------------------------------------------------------------
extern "C" void kernel_launch(void* const* d_in, const int* in_sizes, int n_in,
                              void* d_out, int out_size, void* d_ws, size_t ws_size,
                              hipStream_t stream) {
    (void)in_sizes; (void)n_in; (void)out_size; (void)ws_size;
    constexpr int B = 16, Qn = 1024, Tn = 1024;
    const float* preds   = (const float*)d_in[0];
    const float* targets = (const float*)d_in[1];
    const float* filt    = (const float*)d_in[2];

    // workspace layout
    float* cost = (float*)d_ws;                      // 64 MB
    const size_t ce = (size_t)B * Qn * Tn;
    float* u  = cost + ce;                           // B*Q
    float* v  = u + (size_t)B * Qn;                  // B*T
    float* x2 = v + (size_t)B * Tn;                  // B*Q
    float* y2 = x2 + (size_t)B * Qn;                 // B*T
    unsigned* gmaxBits = (unsigned*)(y2 + (size_t)B * Tn);

    float* outIdx = (float*)d_out;                   // B*Q (argmax as float)
    float* outP   = outIdx + (size_t)B * Qn;         // B*Q*T

    init_k<<<(B * Tn + 255) / 256, 256, 0, stream>>>(v, gmaxBits, B * Tn);
    norms_k<<<B * Qn, 256, 0, stream>>>(preds, x2);
    norms_k<<<B * Tn, 256, 0, stream>>>(targets, y2);
    cost_wmma_k<<<dim3(Tn / 128, Qn / 128, B), 256, 0, stream>>>(
        preds, targets, x2, y2, cost, gmaxBits);
    for (int it = 0; it < 5; ++it) {
        u_update_k<<<dim3(Qn, B), 256, 0, stream>>>(cost, v, filt, gmaxBits, u);
        v_update_k<<<dim3(Tn / 64, B), 256, 0, stream>>>(cost, u, filt, gmaxBits, v);
    }
    p_out_k<<<dim3(Qn, B), 256, 0, stream>>>(cost, u, v, filt, gmaxBits, outIdx, outP);
}
